// SS2D_tiny_8512625181078
// MI455X (gfx1250) — matrix operand order repaired and checked by reading, not compiled, not run on hardware
//
#include <hip/hip_runtime.h>
#include <hip/hip_bf16.h>
#include <math.h>

// ---------------- problem constants (SS2D tiny) ----------------
constexpr int cB = 2;            // batch
constexpr int cC = 96;           // model dim
constexpr int cH = 56, cW = 56;  // spatial
constexpr int cL = cH * cW;      // 3136
constexpr int cD = 192;          // d_inner
constexpr int cN = 16;           // d_state
constexpr int cK = 4;            // scan directions
constexpr int cR = 6;            // dt_rank
constexpr int cXD = cR + 2 * cN; // 38 = x_dbl channels

typedef __attribute__((ext_vector_type(16))) __bf16 v16bf;
typedef __attribute__((ext_vector_type(8)))  float  v8f;

// direction index map: memory index for scan position t in direction k
__device__ __forceinline__ int scan_map(int k, int t) {
  int tt = (k >= 2) ? (cL - 1 - t) : t;
  if (k & 1) tt = (tt % cH) * cW + (tt / cH);  // transpose (H==W==56)
  return tt;
}

__device__ __forceinline__ float softplusf(float x) {
  return (x > 20.f) ? x : log1pf(expf(x));
}

// =====================================================================
// Kernel 1: in_proj GEMM  Xin[b,d,l] = sum_c W[d,c] * x[b,c,l]
// one wave per 16x16 output tile, K=96 in 3 bf16 WMMAs
// =====================================================================
__global__ void k_inproj(const float* __restrict__ x, const float* __restrict__ w,
                         float* __restrict__ xin) {
  const int lane = threadIdx.x;
  const int l0 = blockIdx.x * 16;
  const int m0 = blockIdx.y * 16;
  const int b  = blockIdx.z;
  const float* xb = x + (size_t)b * cC * cL;
  const int mrow = m0 + (lane & 15);
  const int ncol = l0 + (lane & 15);
  v8f acc = {};
  for (int kk = 0; kk < cC; kk += 32) {
    v16bf a, bm;
    const int ka = kk + ((lane & 16) ? 8 : 0);
    const int kb = kk + ((lane & 16) ? 8 : 0);
#pragma unroll
    for (int i = 0; i < 16; ++i) {
      a[i]  = (__bf16)w[mrow * cC + ka + (i < 8 ? i : i + 8)];
      bm[i] = (__bf16)xb[(size_t)(kb + (i < 8 ? i : i + 8)) * cL + ncol];
    }
    acc = __builtin_amdgcn_wmma_f32_16x16x32_bf16(false, a, false, bm,
                                                  (short)0, acc, false, false);
  }
  float* o = xin + (size_t)b * cD * cL;
#pragma unroll
  for (int j = 0; j < 8; ++j)
    o[(size_t)(m0 + j + ((lane & 16) ? 8 : 0)) * cL + ncol] = acc[j];
}

// =====================================================================
// Kernel 2: depthwise 3x3 conv + bias + SiLU  ->  U[b,d,l]
// =====================================================================
__global__ void k_conv(const float* __restrict__ xin, const float* __restrict__ cw,
                       const float* __restrict__ cb, float* __restrict__ U) {
  const int idx = blockIdx.x * blockDim.x + threadIdx.x;
  if (idx >= cB * cD * cL) return;
  const int l = idx % cL;
  const int d = (idx / cL) % cD;
  const int b = idx / (cL * cD);
  const int h = l / cW, w = l % cW;
  const float* src = xin + ((size_t)b * cD + d) * cL;
  float acc = cb[d];
#pragma unroll
  for (int dy = 0; dy < 3; ++dy) {
#pragma unroll
    for (int dx = 0; dx < 3; ++dx) {
      const int hh = h + dy - 1, ww = w + dx - 1;
      if (hh >= 0 && hh < cH && ww >= 0 && ww < cW)
        acc = fmaf(src[hh * cW + ww], cw[d * 9 + dy * 3 + dx], acc);
    }
  }
  U[idx] = acc / (1.f + expf(-acc));   // SiLU
}

// =====================================================================
// Kernel 3: x_dbl GEMM  Xdbl[b,k,c,t] = sum_d Wp[k,d,c] * U[b,d,map_k(t)]
// M=38 padded to 48 (3 tiles); cross-scan applied inside B loader.
// =====================================================================
__global__ void k_xdbl(const float* __restrict__ U, const float* __restrict__ xw,
                       float* __restrict__ xdbl) {
  const int lane = threadIdx.x;
  const int l0 = blockIdx.x * 16;
  const int m0 = blockIdx.y * 16;
  const int bk = blockIdx.z;               // b*K + kdir
  const int kdir = bk % cK, b = bk / cK;
  const float* Ub = U + (size_t)b * cD * cL;
  const float* wk = xw + (size_t)kdir * cD * cXD;
  const int mrow = m0 + (lane & 15);
  const int ncol = l0 + (lane & 15);
  const int lm = scan_map(kdir, ncol);     // memory column for this scan position
  v8f acc = {};
  for (int kk = 0; kk < cD; kk += 32) {
    v16bf a, bm;
    const int ka = kk + ((lane & 16) ? 8 : 0);
    const int kb = kk + ((lane & 16) ? 8 : 0);
#pragma unroll
    for (int i = 0; i < 16; ++i) {
      const int kA = ka + (i < 8 ? i : i + 8);
      a[i]  = (mrow < cXD) ? (__bf16)wk[(size_t)kA * cXD + mrow] : (__bf16)0.0f;
      bm[i] = (__bf16)Ub[(size_t)(kb + (i < 8 ? i : i + 8)) * cL + lm];
    }
    acc = __builtin_amdgcn_wmma_f32_16x16x32_bf16(false, a, false, bm,
                                                  (short)0, acc, false, false);
  }
  float* o = xdbl + (size_t)bk * cXD * cL;
#pragma unroll
  for (int j = 0; j < 8; ++j) {
    const int mj = m0 + j + ((lane & 16) ? 8 : 0);
    if (mj < cXD) o[(size_t)mj * cL + ncol] = acc[j];
  }
}

// =====================================================================
// Kernel 4: dt GEMM (K=6 zero-padded to 32, single WMMA) + bias + softplus
// Delta[b,k,d,t] = softplus( sum_r dtw[k,d,r]*Xdbl[b,k,r,t] + dtb[k,d] )
// =====================================================================
__global__ void k_delta(const float* __restrict__ xdbl, const float* __restrict__ dtw,
                        const float* __restrict__ dtb, float* __restrict__ delta) {
  const int lane = threadIdx.x;
  const int l0 = blockIdx.x * 16;
  const int m0 = blockIdx.y * 16;
  const int bk = blockIdx.z;
  const int kdir = bk % cK;
  const float* xb = xdbl + (size_t)bk * cXD * cL;
  const float* wk = dtw + (size_t)kdir * cD * cR;
  const int mrow = m0 + (lane & 15);
  const int ncol = l0 + (lane & 15);
  v8f acc = {};
  {
    v16bf a, bm;
    const int ka = (lane & 16) ? 8 : 0;
    const int kb = (lane & 16) ? 8 : 0;
#pragma unroll
    for (int i = 0; i < 16; ++i) {
      const int kA = ka + (i < 8 ? i : i + 8);
      const int kB = kb + (i < 8 ? i : i + 8);
      a[i]  = (kA < cR) ? (__bf16)wk[mrow * cR + kA] : (__bf16)0.0f;
      bm[i] = (kB < cR) ? (__bf16)xb[(size_t)kB * cL + ncol] : (__bf16)0.0f;
    }
    acc = __builtin_amdgcn_wmma_f32_16x16x32_bf16(false, a, false, bm,
                                                  (short)0, acc, false, false);
  }
  float* o = delta + (size_t)bk * cD * cL;
#pragma unroll
  for (int j = 0; j < 8; ++j) {
    const int mj = m0 + j + ((lane & 16) ? 8 : 0);
    o[(size_t)mj * cL + ncol] = softplusf(acc[j] + dtb[kdir * cD + mj]);
  }
}

// =====================================================================
// Kernel 5: selective scan. One wave per (b,k,d-pair); lane = (d,n).
// h_n(t) = exp(delta*A_n) h_n(t-1) + delta*B_n(t)*u(t);  y = sum_n h_n*C_n + D*u
// State lives in registers for the whole L=3136 sequence.
// =====================================================================
__global__ void k_scan(const float* __restrict__ U, const float* __restrict__ delta,
                       const float* __restrict__ xdbl, const float* __restrict__ Alog,
                       const float* __restrict__ Dsk, float* __restrict__ ys) {
  const int lane = threadIdx.x;
  const int pair = blockIdx.x % (cD / 2);
  const int kdir = (blockIdx.x / (cD / 2)) % cK;
  const int b    = blockIdx.x / ((cD / 2) * cK);
  const int d = pair * 2 + (lane >> 4);
  const int n = lane & 15;
  const float An = -expf(Alog[((size_t)kdir * cD + d) * cN + n]);
  const float Dv = Dsk[kdir * cD + d];
  const int bk = b * cK + kdir;
  const float* Ub = U     + ((size_t)b  * cD + d) * cL;
  const float* dl = delta + ((size_t)bk * cD + d) * cL;
  const float* Bp = xdbl  + (size_t)bk * cXD * cL + (size_t)(cR + n) * cL;
  const float* Cp = Bp + (size_t)cN * cL;
  float*       yo = ys    + ((size_t)bk * cD + d) * cL;
  float h = 0.f;
  for (int t = 0; t < cL; ++t) {
    const int lm = scan_map(kdir, t);
    const float u  = Ub[lm];
    const float dt = dl[t];
    const float Bn = Bp[t];
    const float Cn = Cp[t];
    h = expf(dt * An) * h + dt * Bn * u;
    float c = h * Cn;
    c += __shfl_xor(c, 8, 16);
    c += __shfl_xor(c, 4, 16);
    c += __shfl_xor(c, 2, 16);
    c += __shfl_xor(c, 1, 16);
    if (n == 0) yo[t] = c + Dv * u;
  }
}

// =====================================================================
// Kernel 6: cross-merge (inverse scans, sum of 4 directions) + exact GELU
// =====================================================================
__global__ void k_merge(const float* __restrict__ ys, float* __restrict__ G) {
  const int idx = blockIdx.x * blockDim.x + threadIdx.x;
  if (idx >= cB * cD * cL) return;
  const int l = idx % cL;
  const int d = (idx / cL) % cD;
  const int b = idx / (cL * cD);
  const int h = l / cW, w = l % cW;
  const int t1 = w * cH + h;                 // inverse transpose position
  const float* base = ys + (size_t)b * cK * cD * cL + (size_t)d * cL;
  const size_t kstr = (size_t)cD * cL;
  float y = base[0 * kstr + l]
          + base[1 * kstr + t1]
          + base[2 * kstr + (cL - 1 - l)]
          + base[3 * kstr + (cL - 1 - t1)];
  G[idx] = 0.5f * y * (1.f + erff(y * 0.7071067811865476f));  // exact GELU
}

// =====================================================================
// Kernel 7: out_proj GEMM  out[b,c,l] = sum_d Wo[c,d] * G[b,d,l]
// =====================================================================
__global__ void k_outproj(const float* __restrict__ G, const float* __restrict__ w,
                          float* __restrict__ out) {
  const int lane = threadIdx.x;
  const int l0 = blockIdx.x * 16;
  const int m0 = blockIdx.y * 16;
  const int b  = blockIdx.z;
  const float* Gb = G + (size_t)b * cD * cL;
  const int mrow = m0 + (lane & 15);
  const int ncol = l0 + (lane & 15);
  v8f acc = {};
  for (int kk = 0; kk < cD; kk += 32) {
    v16bf a, bm;
    const int ka = kk + ((lane & 16) ? 8 : 0);
    const int kb = kk + ((lane & 16) ? 8 : 0);
#pragma unroll
    for (int i = 0; i < 16; ++i) {
      a[i]  = (__bf16)w[mrow * cD + ka + (i < 8 ? i : i + 8)];
      bm[i] = (__bf16)Gb[(size_t)(kb + (i < 8 ? i : i + 8)) * cL + ncol];
    }
    acc = __builtin_amdgcn_wmma_f32_16x16x32_bf16(false, a, false, bm,
                                                  (short)0, acc, false, false);
  }
  float* o = out + (size_t)b * cC * cL;
#pragma unroll
  for (int j = 0; j < 8; ++j)
    o[(size_t)(m0 + j + ((lane & 16) ? 8 : 0)) * cL + ncol] = acc[j];
}

// =====================================================================
extern "C" void kernel_launch(void* const* d_in, const int* in_sizes, int n_in,
                              void* d_out, int out_size, void* d_ws, size_t ws_size,
                              hipStream_t stream) {
  const float* x    = (const float*)d_in[0];
  const float* ipw  = (const float*)d_in[1];
  const float* cw   = (const float*)d_in[2];
  const float* cb   = (const float*)d_in[3];
  const float* xpw  = (const float*)d_in[4];
  const float* dtw  = (const float*)d_in[5];
  const float* dtb  = (const float*)d_in[6];
  const float* alog = (const float*)d_in[7];
  const float* dsp  = (const float*)d_in[8];
  const float* opw  = (const float*)d_in[9];
  (void)in_sizes; (void)n_in; (void)out_size; (void)ws_size;

  float* ws   = (float*)d_ws;
  float* Xin  = ws;                                    // (B,D,L)
  float* U    = Xin  + (size_t)cB * cD * cL;           // (B,D,L)
  float* Xdbl = U    + (size_t)cB * cD * cL;           // (B,K,38,L)
  float* Dlt  = Xdbl + (size_t)cB * cK * cXD * cL;     // (B,K,D,L)
  float* Ys   = Dlt  + (size_t)cB * cK * cD * cL;      // (B,K,D,L)
  float* G    = Ys   + (size_t)cB * cK * cD * cL;      // (B,D,L)

  const dim3 wave(32);
  k_inproj <<<dim3(cL / 16, cD / 16, cB),      wave, 0, stream>>>(x, ipw, Xin);
  const int nelem = cB * cD * cL;
  k_conv   <<<(nelem + 255) / 256, 256, 0, stream>>>(Xin, cw, cb, U);
  k_xdbl   <<<dim3(cL / 16, 3, cB * cK),       wave, 0, stream>>>(U, xpw, Xdbl);
  k_delta  <<<dim3(cL / 16, cD / 16, cB * cK), wave, 0, stream>>>(Xdbl, dtw, dtb, Dlt);
  k_scan   <<<dim3(cB * cK * (cD / 2)),        wave, 0, stream>>>(U, Dlt, Xdbl, alog, dsp, Ys);
  k_merge  <<<(nelem + 255) / 256, 256, 0, stream>>>(Ys, G);
  k_outproj<<<dim3(cL / 16, cC / 16, cB),      wave, 0, stream>>>(G, opw, (float*)d_out);
}
